// GlobalAttention_56100862820999
// MI455X (gfx1250) — matrix operand order repaired and checked by reading, not compiled, not run on hardware
//
#include <hip/hip_runtime.h>

typedef __attribute__((ext_vector_type(16))) _Float16     v16h;
typedef __attribute__((ext_vector_type(8)))  float        v8f;
typedef __attribute__((ext_vector_type(4)))  float        v4f;
typedef __attribute__((ext_vector_type(2)))  float        v2f;
typedef __attribute__((ext_vector_type(4)))  unsigned int v4u;
typedef __attribute__((ext_vector_type(8)))  int          v8i;
typedef __attribute__((ext_vector_type(4)))  int          v4i;

constexpr int Bc  = 2;
constexpr int Tc  = 4096;
constexpr int Cc  = 1024;
constexpr int Hc  = 16;
constexpr int Wc  = 8;
constexpr int DHc = 64;
constexpr int Nc  = Tc / Wc;     // 512 window-sequence length
constexpr int BWc = Bc * Wc;     // 16 effective batches
constexpr int Mr  = Bc * Tc;     // 8192 token rows
constexpr int ASTRIDE = 48;      // LDS row stride in halves (64B row + 32B TDM pad)

__device__ __forceinline__ v8f wmma_f16(v16h a, v16h b, v8f c) {
  // D = A(16x32) * B(32x16) + C, f32 accumulate
  return __builtin_amdgcn_wmma_f32_16x16x32_f16(false, a, false, b, (short)0, c, false, false);
}

// 16 contiguous halves -> B-fragment (lane supplies K=(l>>4)*16 + 0..15)
__device__ __forceinline__ v16h load16h(const _Float16* p) {
  union { float4 f[2]; v16h v; } u;
  u.f[0] = *reinterpret_cast<const float4*>(p);
  u.f[1] = *reinterpret_cast<const float4*>(p + 8);
  return u.v;
}

// Two separated 8-half chunks -> A-fragment (callers: K = hi*16+{0..7}, hi*16+8+{0..7})
__device__ __forceinline__ v16h load8x2h(const _Float16* p0, const _Float16* p1) {
  union { float4 f[2]; v16h v; } u;
  u.f[0] = *reinterpret_cast<const float4*>(p0);
  u.f[1] = *reinterpret_cast<const float4*>(p1);
  return u.v;
}

// ---------------------------------------------------------------------------
// One-shot f32 -> f16 conversion (x and the four weight matrices).
// NT loads: the f32 originals are never touched again.
// ---------------------------------------------------------------------------
__global__ __launch_bounds__(256) void cvt_f16(const float* __restrict__ src,
                                               _Float16* __restrict__ dst, int n4) {
  const int i = blockIdx.x * 256 + threadIdx.x;
  if (i < n4) {
    const v4f v = __builtin_nontemporal_load(reinterpret_cast<const v4f*>(src) + i);
    union { _Float16 h[4]; v2f f; } u;
    u.h[0] = (_Float16)v.x; u.h[1] = (_Float16)v.y;
    u.h[2] = (_Float16)v.z; u.h[3] = (_Float16)v.w;
    *(reinterpret_cast<v2f*>(dst) + i) = u.f;
  }
}

// ---------------------------------------------------------------------------
// GEMM: Y[m,o] = sum_c A[m,c] * Wgt[o,c] + bias[o]   (A, Wgt pre-converted f16)
// Block: 64(M) x 128(N=2 heads), 8 waves = 2(M) x 4(N); wave = 32x(16+16) with its
// two 16-col tiles 32 apart inside a head so RoPE (d, d+32) pairs stay in-wave.
// A tile (64x32 f16) is staged global->LDS by the Tensor Data Mover, double
// buffered; D# pads 32B after every 64B row to reproduce the ASTRIDE=48 layout.
// MODE 0: q/k projection + RoPE, store f16 [BW,H,N,DH]
// MODE 2: v projection, store f16 transposed [BW,H,DH,N]
// MODE 3: out-projection, NT-store f32 to d_out with window un-permute
// ---------------------------------------------------------------------------
template<int MODE>
__global__ __launch_bounds__(256) void gemm_wmma(
    const _Float16* __restrict__ Ain, const _Float16* __restrict__ Wgt,
    const float* __restrict__ bias, _Float16* __restrict__ outH,
    float* __restrict__ outF)
{
  __shared__ __attribute__((aligned(16))) _Float16 As[2][64 * ASTRIDE];  // 2 x 6KB

  const int tid  = threadIdx.x;
  const int wv   = tid >> 5;
  const int lane = tid & 31;
  const int hi   = lane >> 4, lo = lane & 15;
  const int mw   = wv >> 2, nw = wv & 3;
  const int mB   = (int)(blockIdx.x >> 3) * 64;
  const int nB   = (int)(blockIdx.x & 7) * 128;
  const int headB = nw >> 1, off = (nw & 1) * 16;
  const int c0 = nB + headB * 64 + off;   // first 16-col tile
  const int c1 = c0 + 32;                 // paired tile (+32 within head)

  const bool tdmWave = (__builtin_amdgcn_readfirstlane(wv) == 0);  // uniform branch
  const unsigned ldsOff[2] = { (unsigned)(uintptr_t)&As[0][0],
                               (unsigned)(uintptr_t)&As[1][0] };
  const unsigned long long gbase =
      (unsigned long long)(uintptr_t)Ain + (unsigned long long)mB * Cc * 2ull;

  // Issue one TDM load of the 64x32 f16 A tile at column k0 into LDS buffer.
  auto issue_tdm = [&](int k0, unsigned lds) {
    const unsigned long long ga = gbase + (unsigned long long)k0 * 2ull;
    v4u g0;
    g0[0] = 1u;                                        // count=1, user descriptor
    g0[1] = lds;                                       // lds_addr
    g0[2] = (unsigned)(ga & 0xFFFFFFFFull);            // global_addr[31:0]
    g0[3] = (unsigned)((ga >> 32) & 0x01FFFFFFull)     // global_addr[56:32]
          | 0x80000000u;                               // type=2 ("image")
    v8i g1;
    g1[0] = 0x0ED10000;        // wg_mask=0 | data_size=2B | pad_en, intvl=64B, pad=32B
    g1[1] = (int)(1024u << 16);        // tensor_dim0 = 1024 (bits 79:48)
    g1[2] = (int)(8192u << 16);        // tensor_dim1 = 8192 (bits 111:80)
    g1[3] = (int)(32u << 16);          // tile_dim0 = 32 (bits 127:112)
    g1[4] = 64;                        // tile_dim1 = 64, tile_dim2 = 0
    g1[5] = 1024;                      // tensor_dim0_stride = 1024
    g1[6] = 0;
    g1[7] = 0;
    const v4i gz4 = {0, 0, 0, 0};
    const v8i gz8 = {0, 0, 0, 0, 0, 0, 0, 0};
    __builtin_amdgcn_tensor_load_to_lds(g0, g1, gz4, gz4, gz8, 0);
  };

  v8f acc00 = {}, acc01 = {}, acc10 = {}, acc11 = {};

  const _Float16* bp0 = Wgt + (size_t)(c0 + lo) * Cc + hi * 16;
  const _Float16* bp1 = Wgt + (size_t)(c1 + lo) * Cc + hi * 16;

  if (tdmWave) {
    issue_tdm(0, ldsOff[0]);
    __builtin_amdgcn_s_wait_tensorcnt(0);
  }
  __syncthreads();

  for (int k0 = 0; k0 < Cc; k0 += 32) {
    const int cur = (k0 >> 5) & 1;
    if (k0 + 32 < Cc && tdmWave) issue_tdm(k0 + 32, ldsOff[cur ^ 1]);

    __builtin_prefetch(bp0 + k0 + 32, 0, 0);     // global_prefetch_b8 next K-step
    __builtin_prefetch(bp1 + k0 + 32, 0, 0);
    const v16h b0 = load16h(bp0 + k0);
    const v16h b1 = load16h(bp1 + k0);
    const _Float16* Ac = &As[cur][0];
    const int r0 = mw * 32 + lo;
    const v16h a0 = load8x2h(&Ac[r0 * ASTRIDE + hi * 16],
                             &Ac[r0 * ASTRIDE + hi * 16 + 8]);
    const v16h a1 = load8x2h(&Ac[(r0 + 16) * ASTRIDE + hi * 16],
                             &Ac[(r0 + 16) * ASTRIDE + hi * 16 + 8]);
    acc00 = wmma_f16(a0, b0, acc00);
    acc01 = wmma_f16(a0, b1, acc01);
    acc10 = wmma_f16(a1, b0, acc10);
    acc11 = wmma_f16(a1, b1, acc11);

    if (tdmWave) __builtin_amdgcn_s_wait_tensorcnt(0);
    __syncthreads();
  }

  const float bias0 = bias[c0 + lo];
  const float bias1 = bias[c1 + lo];

  #pragma unroll
  for (int mt = 0; mt < 2; ++mt) {
    const v8f& A0 = mt ? acc10 : acc00;
    const v8f& A1 = mt ? acc11 : acc01;
    #pragma unroll
    for (int v = 0; v < 8; ++v) {
      const int m  = mB + mw * 32 + mt * 16 + v + hi * 8;   // token row
      const float x0 = A0[v] + bias0;
      const float x1 = A1[v] + bias1;
      if constexpr (MODE == 0) {                 // q/k + RoPE
        const int bb = m >> 12, t = m & (Tc - 1);
        const int n = t >> 3, j = t & 7;
        const int bw = bb * Wc + j;
        const int h  = nB / 64 + headB;
        const int d0 = off + lo;                 // in [0,32)
        // inv_freq[d0] = 10000^(-d0/32)
        const float fr = (float)n * __expf((float)d0 * (-9.210340371976184f / 32.0f));
        float sn, cs;
        __sincosf(fr, &sn, &cs);
        _Float16* dst = outH + ((size_t)(bw * Hc + h) * Nc + n) * DHc;
        dst[d0]      = (_Float16)(x0 * cs - x1 * sn);   // rotate_half: -x[d+32]
        dst[d0 + 32] = (_Float16)(x1 * cs + x0 * sn);   //             +x[d]
      } else if constexpr (MODE == 2) {          // v, transposed store
        const int bb = m >> 12, t = m & (Tc - 1);
        const int n = t >> 3, j = t & 7;
        const int bw = bb * Wc + j;
        const int h  = nB / 64 + headB;
        const int d0 = off + lo;
        _Float16* dst = outH + (size_t)(bw * Hc + h) * DHc * Nc;
        dst[(size_t)d0 * Nc + n]        = (_Float16)x0;
        dst[(size_t)(d0 + 32) * Nc + n] = (_Float16)x1;
      } else {                                   // out-projection, un-window, NT
        const int bw = m >> 9, n = m & (Nc - 1);
        const int bb = bw >> 3, j = bw & 7;
        const int t  = n * Wc + j;
        float* dst = outF + ((size_t)(bb * Tc + t)) * Cc;
        __builtin_nontemporal_store(x0, dst + c0 + lo);
        __builtin_nontemporal_store(x1, dst + c1 + lo);
      }
    }
  }
}

// ---------------------------------------------------------------------------
// Flash attention over N=512 keys per (bw, h). 512 blocks = (bw, h, q-half).
// 8 waves x 32 queries; Q frags resident; 32-key tiles; online softmax with
// intra-half shfl reductions; P transposed C->A layout via per-wave LDS tile.
// ---------------------------------------------------------------------------
__global__ __launch_bounds__(256) void attn_wmma(
    const _Float16* __restrict__ Q, const _Float16* __restrict__ K,
    const _Float16* __restrict__ VT, const unsigned char* __restrict__ pmask,
    _Float16* __restrict__ X2)
{
  __shared__ __attribute__((aligned(16))) _Float16 Pl[8 * 32 * 32];  // 16KB

  const int tid  = threadIdx.x;
  const int wv   = tid >> 5, lane = tid & 31;
  const int hi   = lane >> 4, lo = lane & 15;
  const int bid  = blockIdx.x;
  const int bw   = bid >> 5;
  const int h    = (bid >> 1) & 15;
  const int q0   = (bid & 1) * 256 + wv * 32;

  const _Float16* Qb = Q  + (size_t)(bw * Hc + h) * Nc * DHc;
  const _Float16* Kb = K  + (size_t)(bw * Hc + h) * Nc * DHc;
  const _Float16* Vb = VT + (size_t)(bw * Hc + h) * DHc * Nc;
  const unsigned char* mrow = pmask + (size_t)(bw >> 3) * Tc;
  const int jj = bw & 7;

  // resident Q A-fragments: [mtile][kstep]
  v16h qa[2][2];
  #pragma unroll
  for (int mt = 0; mt < 2; ++mt)
    #pragma unroll
    for (int ks = 0; ks < 2; ++ks) {
      const _Float16* p = Qb + (size_t)(q0 + mt * 16 + lo) * DHc + ks * 32 + hi * 16;
      qa[mt][ks] = load8x2h(p, p + 8);
    }

  v8f o[2][4];
  float mi[2][8], li[2][8];
  #pragma unroll
  for (int mt = 0; mt < 2; ++mt) {
    #pragma unroll
    for (int dt = 0; dt < 4; ++dt) o[mt][dt] = {};
    #pragma unroll
    for (int v = 0; v < 8; ++v) { mi[mt][v] = -1e30f; li[mt][v] = 0.0f; }
  }

  _Float16* Pw = &Pl[wv * 1024];  // private 32x32 f16 transpose tile
  const float scale = 0.125f;     // 1/sqrt(64)

  for (int kt = 0; kt < Nc; kt += 32) {
    v16h kb[2][2];
    #pragma unroll
    for (int nt = 0; nt < 2; ++nt)
      #pragma unroll
      for (int ks = 0; ks < 2; ++ks)
        kb[nt][ks] = load16h(Kb + (size_t)(kt + nt * 16 + lo) * DHc + ks * 32 + hi * 16);

    const bool keep0 = mrow[(kt + lo) * Wc + jj] != 0;
    const bool keep1 = mrow[(kt + 16 + lo) * Wc + jj] != 0;

    #pragma unroll
    for (int mt = 0; mt < 2; ++mt) {
      v8f S0 = {}, S1 = {};
      S0 = wmma_f16(qa[mt][0], kb[0][0], S0);
      S0 = wmma_f16(qa[mt][1], kb[0][1], S0);
      S1 = wmma_f16(qa[mt][0], kb[1][0], S1);
      S1 = wmma_f16(qa[mt][1], kb[1][1], S1);
      #pragma unroll
      for (int v = 0; v < 8; ++v) {
        const float s0 = keep0 ? S0[v] * scale : -1e30f;
        const float s1 = keep1 ? S1[v] * scale : -1e30f;
        float rm = fmaxf(s0, s1);
        #pragma unroll
        for (int d = 1; d < 16; d <<= 1) rm = fmaxf(rm, __shfl_xor(rm, d, 32));
        const float mn   = fmaxf(mi[mt][v], rm);
        const float corr = __expf(mi[mt][v] - mn);
        const float p0   = __expf(s0 - mn);
        const float p1   = __expf(s1 - mn);
        float rs = p0 + p1;
        #pragma unroll
        for (int d = 1; d < 16; d <<= 1) rs += __shfl_xor(rs, d, 32);
        li[mt][v] = li[mt][v] * corr + rs;
        mi[mt][v] = mn;
        #pragma unroll
        for (int dt = 0; dt < 4; ++dt) o[mt][dt][v] *= corr;
        const int prow = mt * 16 + v + hi * 8;         // C-layout row
        Pw[prow * 32 + lo]      = (_Float16)p0;
        Pw[prow * 32 + 16 + lo] = (_Float16)p1;
      }
    }

    asm volatile("s_wait_dscnt 0x0" ::: "memory");      // own-wave LDS RAW fence

    v16h pa[2];
    #pragma unroll
    for (int mt = 0; mt < 2; ++mt) {
      const _Float16* pp = Pw + (mt * 16 + lo) * 32 + hi * 16;
      pa[mt] = load8x2h(pp, pp + 8);                   // A-layout read of P
    }
    v16h vb[4];
    #pragma unroll
    for (int dt = 0; dt < 4; ++dt)
      vb[dt] = load16h(Vb + (size_t)(dt * 16 + lo) * Nc + kt + hi * 16);

    #pragma unroll
    for (int mt = 0; mt < 2; ++mt)
      #pragma unroll
      for (int dt = 0; dt < 4; ++dt)
        o[mt][dt] = wmma_f16(pa[mt], vb[dt], o[mt][dt]);
  }

  #pragma unroll
  for (int mt = 0; mt < 2; ++mt)
    #pragma unroll
    for (int v = 0; v < 8; ++v) {
      const float inv = 1.0f / li[mt][v];
      const int q = q0 + mt * 16 + v + hi * 8;
      _Float16* dst = X2 + ((size_t)(bw * Nc + q)) * Cc + h * DHc;
      #pragma unroll
      for (int dt = 0; dt < 4; ++dt)
        dst[dt * 16 + lo] = (_Float16)(o[mt][dt][v] * inv);
    }
}

extern "C" void kernel_launch(void* const* d_in, const int* in_sizes, int n_in,
                              void* d_out, int out_size, void* d_ws, size_t ws_size,
                              hipStream_t stream) {
  (void)in_sizes; (void)n_in; (void)out_size; (void)ws_size;
  const float* x  = (const float*)d_in[0];
  const unsigned char* pm = (const unsigned char*)d_in[1];  // bool mask [B,T]
  const float* wq = (const float*)d_in[2];
  const float* bq = (const float*)d_in[3];
  const float* wk = (const float*)d_in[4];
  const float* bk = (const float*)d_in[5];
  const float* wv = (const float*)d_in[6];
  const float* bv = (const float*)d_in[7];
  const float* wo = (const float*)d_in[8];
  const float* bo = (const float*)d_in[9];

  const size_t qkvN = (size_t)BWc * Hc * Nc * DHc;   // 8,388,608 f16 elements
  const size_t wN   = (size_t)Cc * Cc;               // 1,048,576
  _Float16* Qb   = (_Float16*)d_ws;
  _Float16* Kb   = Qb + qkvN;
  _Float16* Vt   = Kb + qkvN;
  _Float16* X2   = Vt + qkvN;        // [BW*N, C] f16 attention output
  _Float16* X16  = X2 + qkvN;        // [B*T, C] f16 copy of x
  _Float16* Wq16 = X16 + (size_t)Mr * Cc;
  _Float16* Wk16 = Wq16 + wN;
  _Float16* Wv16 = Wk16 + wN;
  _Float16* Wo16 = Wv16 + wN;

  dim3 blk(256);
  // one-shot f32 -> f16 pre-pass (NT loads; originals never re-read)
  cvt_f16<<<dim3((unsigned)((size_t)Mr * Cc / 4 / 256)), blk, 0, stream>>>(x,  X16,  (int)((size_t)Mr * Cc / 4));
  cvt_f16<<<dim3((unsigned)(wN / 4 / 256)), blk, 0, stream>>>(wq, Wq16, (int)(wN / 4));
  cvt_f16<<<dim3((unsigned)(wN / 4 / 256)), blk, 0, stream>>>(wk, Wk16, (int)(wN / 4));
  cvt_f16<<<dim3((unsigned)(wN / 4 / 256)), blk, 0, stream>>>(wv, Wv16, (int)(wN / 4));
  cvt_f16<<<dim3((unsigned)(wN / 4 / 256)), blk, 0, stream>>>(wo, Wo16, (int)(wN / 4));

  dim3 grid(1024);
  gemm_wmma<0><<<grid, blk, 0, stream>>>(X16, Wq16, bq, Qb, nullptr);
  gemm_wmma<0><<<grid, blk, 0, stream>>>(X16, Wk16, bk, Kb, nullptr);
  gemm_wmma<2><<<grid, blk, 0, stream>>>(X16, Wv16, bv, Vt, nullptr);
  attn_wmma<<<dim3(BWc * Hc * 2), blk, 0, stream>>>(Qb, Kb, Vt, pm, X2);
  gemm_wmma<3><<<grid, blk, 0, stream>>>(X2, Wo16, bo, nullptr, (float*)d_out);
}
